// LCAOInteraction_4140348473501
// MI455X (gfx1250) — matrix operand order repaired and checked by reading, not compiled, not run on hardware
//
#include <hip/hip_runtime.h>
#include <hip/hip_bf16.h>

#define NN   10000
#define EE   200000
#define TT   400000
#define NORB 9
#define HH   128
#define CFDIM 64
#define CC   32

typedef _Float16 v16h __attribute__((ext_vector_type(16)));
typedef _Float16 v8h  __attribute__((ext_vector_type(8)));
typedef _Float16 v4h  __attribute__((ext_vector_type(4)));
typedef float    v8f  __attribute__((ext_vector_type(8)));

// Fast transcendental helpers: v_rcp_f32 / v_rsq_f32 instead of IEEE divide
// sequences (v_div_scale + Newton). Accuracy ~1 ulp, fine here.
__device__ __forceinline__ float siluf(float x) {
  return x * __builtin_amdgcn_rcpf(1.0f + __expf(-x));
}
__device__ __forceinline__ float sigmf(float x) {
  return __builtin_amdgcn_rcpf(1.0f + __expf(-x));
}
// 1 / max(sqrt(ss), 1e-12) as a single v_rsq_f32 with clamped input.
__device__ __forceinline__ float invnorm(float ss) {
  return __builtin_amdgcn_rsqf(fmaxf(ss, 1e-24f));
}

__device__ __forceinline__ float waveSum(float v) {
#pragma unroll
  for (int o = 16; o > 0; o >>= 1) v += __shfl_xor(v, o, 32);
  return v;
}

// ---------------------------------------------------------------------------
// Kernel 1: h = x @ W1^T + b1 ; xh = h[:,0:32], sigxk = sigmoid(h[:,32:64])
// ---------------------------------------------------------------------------
__global__ void k_node_mlp(const float* __restrict__ x,
                           const float* __restrict__ W1,   // (64,128)
                           const float* __restrict__ b1,   // (64)
                           float* __restrict__ xh,
                           float* __restrict__ sigxk) {
  __shared__ float W1T[HH][2 * CC];   // transposed: [k][ch], conflict-free reads
  __shared__ float b1s[2 * CC];
  for (int idx = threadIdx.x; idx < HH * 2 * CC; idx += blockDim.x) {
    int ch = idx / HH, k = idx % HH;
    W1T[k][ch] = W1[idx];
  }
  if (threadIdx.x < 2 * CC) b1s[threadIdx.x] = b1[threadIdx.x];
  __syncthreads();

  int wave = threadIdx.x >> 5, lane = threadIdx.x & 31;
  int n = blockIdx.x * 8 + wave;
  if (n >= NN) return;
  const float4* xr = (const float4*)(x + (size_t)n * HH);
  float ah = b1s[lane], ak = b1s[CC + lane];
#pragma unroll 4
  for (int q = 0; q < HH / 4; ++q) {
    float4 xv = xr[q];
    int k = q * 4;
    ah += xv.x * W1T[k + 0][lane] + xv.y * W1T[k + 1][lane] +
          xv.z * W1T[k + 2][lane] + xv.w * W1T[k + 3][lane];
    ak += xv.x * W1T[k + 0][CC + lane] + xv.y * W1T[k + 1][CC + lane] +
          xv.z * W1T[k + 2][CC + lane] + xv.w * W1T[k + 3][CC + lane];
  }
  xh[(size_t)n * CC + lane]    = ah;
  sigxk[(size_t)n * CC + lane] = sigmf(ak);
}

// ---------------------------------------------------------------------------
// Kernel 2 (WMMA): per 16-row tile of the flattened (E*9, 64) coefficient
// matrix:  c64 = silu( silu(cji) @ W2^T ) @ W3^T ; cji_c = c64[:, :32],
// ckjn = l2norm_rows(c64[:, 32:])
// ---------------------------------------------------------------------------
__global__ void k_edge_coeff(const float* __restrict__ cji,   // (E*9, 64)
                             const float* __restrict__ W2,    // (32, 64)
                             const float* __restrict__ W3,    // (64, 32)
                             float* __restrict__ cji_c,       // (E*9, 32)
                             float* __restrict__ ckjn) {      // (E*9, 32)
  // Per-wave LDS regions; rows padded so 16B vector loads stay aligned and
  // the f32 scratch avoids bank conflicts.
  __shared__ _Float16 sA[8][16][80];  // silu(cji) tile as f16 (row stride 160B)
  __shared__ _Float16 sI[8][16][40];  // intermediate silu tile (row stride 80B)
  __shared__ float    sO[8][16][33];  // raw ckj tile for row-normalization
  __shared__ float    sR[8][16];      // per-row 1/max(norm,eps)

  const int wave = threadIdx.x >> 5;
  const int lane = threadIdx.x & 31;
  const int tile = blockIdx.x * 8 + wave;
  const int numTiles = (EE * NORB) / 16;   // 112500, exact
  if (tile >= numTiles) return;            // wave-uniform; EXEC stays full
  const int r0  = tile * 16;
  const int nlo = lane & 15;
  const int hi  = lane >> 4;

  // --- B fragments (k order shared with A: lane = column, K = hi*16 + i) ---
  v16h b1f[2][2];   // [kstep][ntile] for GEMM1: B[k][n] = W2[n][k]
#pragma unroll
  for (int ks = 0; ks < 2; ++ks)
#pragma unroll
    for (int nt = 0; nt < 2; ++nt) {
      const float* p = W2 + (size_t)(nt * 16 + nlo) * CFDIM + ks * 32 + hi * 16;
      v16h b;
#pragma unroll
      for (int i = 0; i < 16; ++i) b[i] = (_Float16)p[i];
      b1f[ks][nt] = b;
    }
  v16h b2f[4];      // GEMM2: B[k][n] = W3[n][k], K=32 single step
#pragma unroll
  for (int nt = 0; nt < 4; ++nt) {
    const float* p = W3 + (size_t)(nt * 16 + nlo) * CC + hi * 16;
    v16h b;
#pragma unroll
    for (int i = 0; i < 16; ++i) b[i] = (_Float16)p[i];
    b2f[nt] = b;
  }

  // --- stage A tile: silu(cji) -> f16 in LDS ---
#pragma unroll
  for (int q = 0; q < 8; ++q) {
    int idx = lane + 32 * q;            // 256 float4 = 16 rows x 16 quads
    int row = idx >> 4, kq = idx & 15;
    const float4 s = *(const float4*)(cji + (size_t)(r0 + row) * CFDIM + kq * 4);
    v4h h;
    h[0] = (_Float16)siluf(s.x);
    h[1] = (_Float16)siluf(s.y);
    h[2] = (_Float16)siluf(s.z);
    h[3] = (_Float16)siluf(s.w);
    *(v4h*)&sA[wave][row][kq * 4] = h;
  }

  // --- GEMM1: (16x64) x (64x32), 2 K-steps x 2 N-tiles ---
  v8f c1[2] = {};
#pragma unroll
  for (int ks = 0; ks < 2; ++ks) {
    // A in the k order used for B: lane holds row m=nlo; K chunk [hi*16,+16) contiguous
    v8h lo = *(const v8h*)&sA[wave][nlo][ks * 32 + hi * 16];
    v8h up = *(const v8h*)&sA[wave][nlo][ks * 32 + hi * 16 + 8];
    v16h a = __builtin_shufflevector(lo, up, 0, 1, 2, 3, 4, 5, 6, 7,
                                             8, 9, 10, 11, 12, 13, 14, 15);
#pragma unroll
    for (int nt = 0; nt < 2; ++nt)
      c1[nt] = __builtin_amdgcn_wmma_f32_16x16x32_f16(
          false, a, false, b1f[ks][nt], (short)0, c1[nt], false, false);
  }

  // --- silu + repack C (m = v + hi*8, n = nt*16 + nlo) into A2 tile ---
#pragma unroll
  for (int nt = 0; nt < 2; ++nt)
#pragma unroll
    for (int v = 0; v < 8; ++v)
      sI[wave][v + hi * 8][nt * 16 + nlo] = (_Float16)siluf(c1[nt][v]);

  // --- GEMM2: (16x32) x (32x64), 1 K-step x 4 N-tiles ---
  v8h lo2 = *(const v8h*)&sI[wave][nlo][hi * 16];
  v8h up2 = *(const v8h*)&sI[wave][nlo][hi * 16 + 8];
  v16h a2 = __builtin_shufflevector(lo2, up2, 0, 1, 2, 3, 4, 5, 6, 7,
                                              8, 9, 10, 11, 12, 13, 14, 15);
  v8f d2[4] = {};
#pragma unroll
  for (int nt = 0; nt < 4; ++nt)
    d2[nt] = __builtin_amdgcn_wmma_f32_16x16x32_f16(
        false, a2, false, b2f[nt], (short)0, d2[nt], false, false);

  // --- store cji_c (columns 0..31) ---
#pragma unroll
  for (int nt = 0; nt < 2; ++nt)
#pragma unroll
    for (int v = 0; v < 8; ++v)
      cji_c[(size_t)(r0 + v + hi * 8) * CC + nt * 16 + nlo] = d2[nt][v];

  // --- ckj (columns 32..63): row-normalize once here (shared by triplets) ---
#pragma unroll
  for (int nt = 2; nt < 4; ++nt)
#pragma unroll
    for (int v = 0; v < 8; ++v)
      sO[wave][v + hi * 8][(nt - 2) * 16 + nlo] = d2[nt][v];

  if (lane < 16) {
    float ss = 0.f;
#pragma unroll
    for (int c = 0; c < CC; ++c) { float t = sO[wave][lane][c]; ss += t * t; }
    sR[wave][lane] = invnorm(ss);
  }
#pragma unroll
  for (int q = 0; q < 16; ++q) {
    int idx = lane + 32 * q;            // 512 elements cooperative store
    int row = idx >> 5, col = idx & 31;
    ckjn[(size_t)(r0 + row) * CC + col] = sO[wave][row][col] * sR[wave][row];
  }
}

// ---------------------------------------------------------------------------
// Kernel 3: triplet contraction + l2norm + gate + scatter to edges
// ---------------------------------------------------------------------------
__global__ void k_triplet(const float* __restrict__ rb,        // (E,9)
                          const float* __restrict__ cutoff_w,  // (E)
                          const float* __restrict__ shb,       // (T,9)
                          const int* __restrict__ edge_idx_kj,
                          const int* __restrict__ edge_idx_ji,
                          const int* __restrict__ tri_idx_k,
                          const float* __restrict__ ckjn,      // (E*9,32) pre-normalized
                          const float* __restrict__ sigxk,     // (N,32)
                          float* __restrict__ agg) {           // (E,32) pre-zeroed
  int wave = threadIdx.x >> 5, lane = threadIdx.x & 31;
  int t = blockIdx.x * 8 + wave;
  if (t >= TT) return;
  int kj = edge_idx_kj[t];
  const float* crow = ckjn + (size_t)kj * NORB * CC;
  __builtin_prefetch(crow, 0, 3);                 // global_prefetch_b8
  float cw = cutoff_w[kj];
  const float* rbr = rb + (size_t)kj * NORB;
  const float* shr = shb + (size_t)t * NORB;
  float acc = 0.f;
#pragma unroll
  for (int d = 0; d < NORB; ++d) {
    float w = rbr[d] * cw * shr[d];
    acc += w * crow[d * CC + lane];
  }
  float inv = invnorm(waveSum(acc * acc));
  int   kn  = tri_idx_k[t];
  float tw  = acc * inv * sigxk[(size_t)kn * CC + lane];
  atomicAdd(&agg[(size_t)edge_idx_ji[t] * CC + lane], tw);
}

// ---------------------------------------------------------------------------
// Kernel 4: three_lin, coefficient modulation, lcao weight, node_lin, scatter
// ---------------------------------------------------------------------------
__global__ void k_edge_final(const float* __restrict__ agg,
                             const float* __restrict__ W4, const float* __restrict__ b4,
                             const float* __restrict__ W5, const float* __restrict__ b5,
                             const float* __restrict__ W6, const float* __restrict__ b6,
                             const float* __restrict__ rb, const float* __restrict__ cutoff_w,
                             const float* __restrict__ cji_c,
                             const float* __restrict__ xh,
                             const int* __restrict__ idx_i, const int* __restrict__ idx_j,
                             float* __restrict__ accN) {        // (N,32) pre-zeroed
  __shared__ float W4T[CC][CC];       // [k][out]
  __shared__ float W5T[2 * CC][CC];
  __shared__ float W6T[CC][CC];
  __shared__ float b4s[CC], b5s[CC], b6s[CC];
  for (int idx = threadIdx.x; idx < CC * CC; idx += blockDim.x) {
    int r = idx / CC, c = idx % CC;
    W4T[c][r] = W4[idx];
    W6T[c][r] = W6[idx];
  }
  for (int idx = threadIdx.x; idx < CC * 2 * CC; idx += blockDim.x) {
    int r = idx / (2 * CC), c = idx % (2 * CC);
    W5T[c][r] = W5[idx];
  }
  if (threadIdx.x < CC) {
    b4s[threadIdx.x] = b4[threadIdx.x];
    b5s[threadIdx.x] = b5[threadIdx.x];
    b6s[threadIdx.x] = b6[threadIdx.x];
  }
  __syncthreads();

  int wave = threadIdx.x >> 5, lane = threadIdx.x & 31;
  int e = blockIdx.x * 8 + wave;
  if (e >= EE) return;

  // tbw = silu(agg[e]) @ W4^T + b4
  float sag = siluf(agg[(size_t)e * CC + lane]);
  float tb = b4s[lane];
#pragma unroll
  for (int k = 0; k < CC; ++k) tb += __shfl(sag, k, 32) * W4T[k][lane];
  float mod = 1.0f + tb;

  // lcao_w = l2norm( sum_d rb_w[e,d] * l2norm(cji_c[e,d,:] * (1+tbw)) )
  float cw = cutoff_w[e];
  const float* cbase = cji_c + (size_t)e * NORB * CC;
  const float* rbr   = rb + (size_t)e * NORB;
  float lw = 0.f;
#pragma unroll
  for (int d = 0; d < NORB; ++d) {
    float v   = cbase[d * CC + lane] * mod;
    float inv = invnorm(waveSum(v * v));
    lw += (rbr[d] * cw) * v * inv;
  }
  lw *= invnorm(waveSum(lw * lw));

  // nf = silu(silu([xh_i, xh_j]) @ W5^T + b5) @ W6^T + b6
  int i = idx_i[e], j = idx_j[e];
  float si = siluf(xh[(size_t)i * CC + lane]);
  float sj = siluf(xh[(size_t)j * CC + lane]);
  float h1 = b5s[lane];
#pragma unroll
  for (int k = 0; k < CC; ++k) {
    h1 += __shfl(si, k, 32) * W5T[k][lane];
    h1 += __shfl(sj, k, 32) * W5T[CC + k][lane];
  }
  float s1 = siluf(h1);
  float h2 = b6s[lane];
#pragma unroll
  for (int k = 0; k < CC; ++k) h2 += __shfl(s1, k, 32) * W6T[k][lane];

  atomicAdd(&accN[(size_t)i * CC + lane], lw * h2);
}

// ---------------------------------------------------------------------------
// Kernel 5: out = x + accN @ W7^T
// ---------------------------------------------------------------------------
__global__ void k_output(const float* __restrict__ x,
                         const float* __restrict__ accN,
                         const float* __restrict__ W7,   // (128,32)
                         float* __restrict__ out) {
  __shared__ float W7T[CC][HH];       // [c][hh], conflict-free reads
  for (int idx = threadIdx.x; idx < HH * CC; idx += blockDim.x) {
    int r = idx / CC, c = idx % CC;
    W7T[c][r] = W7[idx];
  }
  __syncthreads();
  int wave = threadIdx.x >> 5, lane = threadIdx.x & 31;
  int n = blockIdx.x * 8 + wave;
  if (n >= NN) return;
  float a = accN[(size_t)n * CC + lane];
  float o[4];
#pragma unroll
  for (int jj = 0; jj < 4; ++jj) o[jj] = x[(size_t)n * HH + lane + 32 * jj];
#pragma unroll
  for (int c = 0; c < CC; ++c) {
    float av = __shfl(a, c, 32);
#pragma unroll
    for (int jj = 0; jj < 4; ++jj) o[jj] += av * W7T[c][lane + 32 * jj];
  }
#pragma unroll
  for (int jj = 0; jj < 4; ++jj) out[(size_t)n * HH + lane + 32 * jj] = o[jj];
}

// ---------------------------------------------------------------------------
extern "C" void kernel_launch(void* const* d_in, const int* in_sizes, int n_in,
                              void* d_out, int out_size, void* d_ws, size_t ws_size,
                              hipStream_t stream) {
  const float* x         = (const float*)d_in[0];
  const float* cji       = (const float*)d_in[1];
  const float* cutoff_w  = (const float*)d_in[2];
  const float* rb        = (const float*)d_in[3];
  const float* shb       = (const float*)d_in[4];
  const int*   idx_i     = (const int*)d_in[5];
  const int*   idx_j     = (const int*)d_in[6];
  const int*   tri_idx_k = (const int*)d_in[7];
  const int*   e_kj      = (const int*)d_in[8];
  const int*   e_ji      = (const int*)d_in[9];
  const float* W1 = (const float*)d_in[10];
  const float* b1 = (const float*)d_in[11];
  const float* W2 = (const float*)d_in[12];
  const float* W3 = (const float*)d_in[13];
  const float* W4 = (const float*)d_in[14];
  const float* b4 = (const float*)d_in[15];
  const float* W5 = (const float*)d_in[16];
  const float* b5 = (const float*)d_in[17];
  const float* W6 = (const float*)d_in[18];
  const float* b6 = (const float*)d_in[19];
  const float* W7 = (const float*)d_in[20];
  float* out = (float*)d_out;
  (void)in_sizes; (void)n_in; (void)out_size; (void)ws_size;

  char* ws = (char*)d_ws;
  size_t off = 0;
  auto wsf = [&](size_t bytes) -> float* {
    float* p = (float*)(ws + off);
    off += (bytes + 255) & ~(size_t)255;
    return p;
  };
  float* xh    = wsf((size_t)NN * CC * 4);
  float* sigxk = wsf((size_t)NN * CC * 4);
  float* agg   = wsf((size_t)EE * CC * 4);
  float* accN  = wsf((size_t)NN * CC * 4);
  float* cji_c = wsf((size_t)EE * NORB * CC * 4);
  float* ckjn  = wsf((size_t)EE * NORB * CC * 4);

  hipMemsetAsync(agg,  0, (size_t)EE * CC * 4, stream);
  hipMemsetAsync(accN, 0, (size_t)NN * CC * 4, stream);

  k_node_mlp<<<(NN + 7) / 8, 256, 0, stream>>>(x, W1, b1, xh, sigxk);

  const int numTiles = (EE * NORB) / 16;   // 112500
  k_edge_coeff<<<(numTiles + 7) / 8, 256, 0, stream>>>(cji, W2, W3, cji_c, ckjn);

  k_triplet<<<(TT + 7) / 8, 256, 0, stream>>>(rb, cutoff_w, shb, e_kj, e_ji,
                                              tri_idx_k, ckjn, sigxk, agg);

  k_edge_final<<<(EE + 7) / 8, 256, 0, stream>>>(agg, W4, b4, W5, b5, W6, b6,
                                                 rb, cutoff_w, cji_c, xh,
                                                 idx_i, idx_j, accN);

  k_output<<<(NN + 7) / 8, 256, 0, stream>>>(x, accN, W7, out);
}
